// NeuralRDE_84808424227318
// MI455X (gfx1250) — matrix operand order repaired and checked by reading, not compiled, not run on hardware
//
#include <hip/hip_runtime.h>
#include <hip/hip_bf16.h>

// Neural RDE on MI455X (gfx1250), wave32 + WMMA bf16.
// Persistent kernel: 16 workgroups x 512 threads (16 waves). Each WG owns a
// 16-sample batch tile and runs the whole T=512 recurrence in one launch.
// Weights pre-converted to bf16 in d_ws (~5 MB, L2-resident).
// Activations built directly on raw v_exp_f32/v_log_f32/v_rcp_f32 (TRANS32,
// co-executes with the WMMA pipe); no libm guard code.

#define BB   256
#define TT   512
#define SS   256
#define LL   36
#define DIN  8
#define OUTD 10
#define WW   256

typedef __attribute__((ext_vector_type(16))) __bf16        v16bf;
typedef __attribute__((ext_vector_type(8)))  float         v8f;
typedef __attribute__((ext_vector_type(4)))  unsigned int  v4u;

union HalfPack { v4u u[2]; v16bf v; };

#define LOG2E_F 1.4426950408889634f
#define LN2_F   0.6931471805599453f

__device__ __forceinline__ unsigned short f32_to_bf16_bits(float f) {
  unsigned int u = __float_as_uint(f);
  u += 0x7FFFu + ((u >> 16) & 1u);   // round-to-nearest-even
  return (unsigned short)(u >> 16);
}

// softplus(x) = max(x,0) + ln2 * log2(1 + 2^(-log2e*|x|))
// argument of log2 is in [1,2] -> raw v_log_f32 is safe (no denorm guard).
__device__ __forceinline__ float softplus_f(float x) {
  float e = __builtin_amdgcn_exp2f(-LOG2E_F * fabsf(x));
  return fmaxf(x, 0.0f) + LN2_F * __builtin_amdgcn_logf(1.0f + e);
}

// tanh(x) = 1 - 2 / (2^(2*log2e*x) + 1)   (raw v_exp_f32 + v_rcp_f32)
__device__ __forceinline__ float tanh_fast(float x) {
  float e = __builtin_amdgcn_exp2f((2.0f * LOG2E_F) * x);
  return 1.0f - 2.0f * __builtin_amdgcn_rcpf(e + 1.0f);
}

#define WMMA_BF16(a, b, c) \
  __builtin_amdgcn_wmma_f32_16x16x32_bf16(false, (a), false, (b), (short)0, (c), false, false)

// ---------------- prep: f32 -> bf16 weight conversion ----------------
__global__ void cvt_bf16_kernel(const float* __restrict__ src,
                                unsigned short* __restrict__ dst, int n) {
  int i = blockIdx.x * blockDim.x + threadIdx.x;
  if (i < n) dst[i] = f32_to_bf16_bits(src[i]);
}

__global__ void pad_ro_kernel(const float* __restrict__ roW,
                              unsigned short* __restrict__ dst) {
  int i = blockIdx.x * blockDim.x + threadIdx.x;
  if (i < 16 * SS) {
    int nn = i / SS, k = i - nn * SS;
    dst[i] = (nn < OUTD) ? f32_to_bf16_bits(roW[nn * SS + k]) : (unsigned short)0;
  }
}

// One 16x16 C tile, K=256. A in LDS bf16 [16][256]; B row-major bf16 [N][256].
// Software-pipelined: exactly two (A,B) pairs in flight, two accumulators.
__device__ __forceinline__ v8f wmma_tile_pipe(const unsigned short* __restrict__ Bmat,
                                              int n0,
                                              const unsigned short* Amat,
                                              int lane) {
  const int lm = lane & 15;
  const int lh = lane >> 4;
  const unsigned short* arow = Amat + lm * 256 + lh * 8;                  // A row = lane&15
  const unsigned short* brow = Bmat + (size_t)(n0 + lm) * 256 + lh * 8;   // B col = lane&15
  v8f c0 = {}, c1 = {};
#pragma unroll 1
  for (int kb = 0; kb < 256; kb += 64) {
    HalfPack a0, a1, b0, b1;
    b0.u[0] = *(const v4u*)(brow + kb);
    b0.u[1] = *(const v4u*)(brow + kb + 16);
    b1.u[0] = *(const v4u*)(brow + kb + 32);
    b1.u[1] = *(const v4u*)(brow + kb + 48);
    a0.u[0] = *(const v4u*)(arow + kb);
    a0.u[1] = *(const v4u*)(arow + kb + 16);
    a1.u[0] = *(const v4u*)(arow + kb + 32);
    a1.u[1] = *(const v4u*)(arow + kb + 48);
    c0 = WMMA_BF16(a0.v, b0.v, c0);
    c1 = WMMA_BF16(a1.v, b1.v, c1);
  }
  return c0 + c1;
}

__global__ __launch_bounds__(512, 1)
void rde_kernel(const float* __restrict__ x0,
                const float* __restrict__ logsigs,
                const float* __restrict__ iW1, const float* __restrict__ ib1,
                const float* __restrict__ iW2, const float* __restrict__ ib2,
                const float* __restrict__ iW3, const float* __restrict__ ib3,
                const float* __restrict__ vb1, const float* __restrict__ vb2,
                const float* __restrict__ vb3,
                const float* __restrict__ roW, const float* __restrict__ rob,
                const unsigned short* __restrict__ W1bf,
                const unsigned short* __restrict__ W2bf,
                const unsigned short* __restrict__ W3bf,
                const unsigned short* __restrict__ robf,
                float* __restrict__ out) {
  __shared__ __align__(16) unsigned short h_bf[16 * 256];   // hidden, bf16 (WMMA A)
  __shared__ __align__(16) unsigned short z1_bf[16 * 256];
  __shared__ __align__(16) unsigned short z2_bf[16 * 256];
  __shared__ __align__(16) float h_f[16 * 256];             // hidden, f32 master
  __shared__ __align__(16) float hacc[16 * 256];            // per-wave 16x16 accum
  __shared__ __align__(16) float ls_s[16 * LL];             // logsig tile for step t

  const int tid   = threadIdx.x;
  const int w     = tid >> 5;     // wave id 0..15
  const int lane  = tid & 31;
  const int lm    = lane & 15;
  const int lh    = lane >> 4;
  const int bbase = blockIdx.x * 16;

  // ---------------- init-condition MLP (one time, VALU; tiny) ----------------
  for (int i = tid; i < 16 * 256; i += 512) {               // layer 1: D_IN=8 -> W
    int b = i >> 8, j = i & 255;
    float s = ib1[j];
    const float* xr = x0 + (bbase + b) * DIN;
    const float* wr = iW1 + j * DIN;
#pragma unroll
    for (int k = 0; k < DIN; k++) s += xr[k] * wr[k];
    h_f[i] = softplus_f(s);
  }
  __syncthreads();
  for (int i = tid; i < 16 * 256; i += 512) {               // layer 2: W -> W
    int b = i >> 8, j = i & 255;
    float s = ib2[j];
    const float* wr = iW2 + j * 256;
    const float* ar = h_f + b * 256;
    for (int k = 0; k < 256; k++) s += ar[k] * wr[k];
    hacc[i] = softplus_f(s);
  }
  __syncthreads();
  for (int i = tid; i < 16 * 256; i += 512) {               // layer 3: W -> S (h0)
    int b = i >> 8, j = i & 255;
    float s = ib3[j];
    const float* wr = iW3 + j * 256;
    const float* ar = hacc + b * 256;
    for (int k = 0; k < 256; k++) s += ar[k] * wr[k];
    h_f[i]  = s;
    h_bf[i] = f32_to_bf16_bits(s);
  }
  __syncthreads();
  for (int i = tid; i < 16 * OUTD; i += 512) {              // readout at t=0
    int b = i / OUTD, o = i - b * OUTD;
    float s = rob[o];
    const float* wr = roW + o * 256;
    const float* hr = h_f + b * 256;
    for (int k = 0; k < 256; k++) s += hr[k] * wr[k];
    out[(size_t)(bbase + b) * (TT + 1) * OUTD + o] = s;
  }
  __syncthreads();

  // ---------------- sequential scan over T steps ----------------
  for (int t = 0; t < TT; t++) {
    // stage this step's logsig tile: ls[b, t, :]
    for (int i = tid; i < 16 * LL; i += 512) {
      int b = i / LL, l = i - b * LL;
      ls_s[i] = logsigs[((size_t)(bbase + b) * TT + t) * LL + l];
    }
    __syncthreads();

    // GEMM1: z1 = softplus(h @ vf_W1^T + b1); wave w owns columns [16w,16w+16)
    {
      const int n0 = w * 16;
      v8f c = wmma_tile_pipe(W1bf, n0, h_bf, lane);
      float bias = vb1[n0 + lm];
#pragma unroll
      for (int r = 0; r < 8; r++)
        z1_bf[(lh * 8 + r) * 256 + n0 + lm] =
            f32_to_bf16_bits(softplus_f(c[r] + bias));
    }
    __syncthreads();

    // GEMM2: z2 = softplus(z1 @ vf_W2^T + b2)
    {
      const int n0 = w * 16;
      v8f c = wmma_tile_pipe(W2bf, n0, z1_bf, lane);
      float bias = vb2[n0 + lm];
#pragma unroll
      for (int r = 0; r < 8; r++)
        z2_bf[(lh * 8 + r) * 256 + n0 + lm] =
            f32_to_bf16_bits(softplus_f(c[r] + bias));
    }
    __syncthreads();

    // zero this wave's h-delta accumulator (wave-local, LDS in-order)
#pragma unroll
    for (int j = 0; j < 8; j++) hacc[w * 256 + lane * 8 + j] = 0.0f;

    // ---- GEMM3 + einsum ----
    // A (= z2 rows, full K=256) kept register-resident across all 36 N-tiles:
    // eight named packs -> constant indices -> no scratch.
    {
      const unsigned short* arow = z2_bf + lm * 256 + lh * 8;
      HalfPack A0, A1, A2, A3, A4, A5, A6, A7;
      A0.u[0] = *(const v4u*)(arow +   0); A0.u[1] = *(const v4u*)(arow +  16);
      A1.u[0] = *(const v4u*)(arow +  32); A1.u[1] = *(const v4u*)(arow +  48);
      A2.u[0] = *(const v4u*)(arow +  64); A2.u[1] = *(const v4u*)(arow +  80);
      A3.u[0] = *(const v4u*)(arow +  96); A3.u[1] = *(const v4u*)(arow + 112);
      A4.u[0] = *(const v4u*)(arow + 128); A4.u[1] = *(const v4u*)(arow + 144);
      A5.u[0] = *(const v4u*)(arow + 160); A5.u[1] = *(const v4u*)(arow + 176);
      A6.u[0] = *(const v4u*)(arow + 192); A6.u[1] = *(const v4u*)(arow + 208);
      A7.u[0] = *(const v4u*)(arow + 224); A7.u[1] = *(const v4u*)(arow + 240);

#pragma unroll 1
      for (int nt = 0; nt < 36; nt++) {
        const int n0 = w * 576 + nt * 16;
        const unsigned short* brow = W3bf + (size_t)(n0 + lm) * 256 + lh * 8;
        v8f c0 = {}, c1 = {};
        HalfPack b0, b1, b2, b3, b4, b5, b6, b7;
        b0.u[0] = *(const v4u*)(brow +   0); b0.u[1] = *(const v4u*)(brow +  16);
        b1.u[0] = *(const v4u*)(brow +  32); b1.u[1] = *(const v4u*)(brow +  48);
        b2.u[0] = *(const v4u*)(brow +  64); b2.u[1] = *(const v4u*)(brow +  80);
        b3.u[0] = *(const v4u*)(brow +  96); b3.u[1] = *(const v4u*)(brow + 112);
        c0 = WMMA_BF16(A0.v, b0.v, c0);
        c1 = WMMA_BF16(A1.v, b1.v, c1);
        b4.u[0] = *(const v4u*)(brow + 128); b4.u[1] = *(const v4u*)(brow + 144);
        b5.u[0] = *(const v4u*)(brow + 160); b5.u[1] = *(const v4u*)(brow + 176);
        c0 = WMMA_BF16(A2.v, b2.v, c0);
        c1 = WMMA_BF16(A3.v, b3.v, c1);
        b6.u[0] = *(const v4u*)(brow + 192); b6.u[1] = *(const v4u*)(brow + 208);
        b7.u[0] = *(const v4u*)(brow + 224); b7.u[1] = *(const v4u*)(brow + 240);
        c0 = WMMA_BF16(A4.v, b4.v, c0);
        c1 = WMMA_BF16(A5.v, b5.v, c1);
        c0 = WMMA_BF16(A6.v, b6.v, c0);
        c1 = WMMA_BF16(A7.v, b7.v, c1);
        v8f c = c0 + c1;

        const unsigned int n = (unsigned int)(n0 + lm);
        const unsigned int s = n / 36u;           // state index
        const unsigned int l = n - s * 36u;       // logsig index
        const float bias = vb3[n];
        const int sc = (int)s - w * 16;           // 0..15 inside wave tile
#pragma unroll
        for (int r = 0; r < 8; r++) {
          int m = lh * 8 + r;                     // batch row in tile
          float val = tanh_fast(c[r] + bias) * ls_s[m * LL + (int)l];
          atomicAdd(&hacc[w * 256 + m * 16 + sc], val);   // ds_add_f32, wave-local
        }
      }
    }

    // h <- h + delta  (wave w owns s-range [16w,16w+16) for all 16 batch rows)
#pragma unroll
    for (int j = 0; j < 8; j++) {
      int i = lane * 8 + j;
      int m = i >> 4, sc = i & 15;
      int idx = m * 256 + w * 16 + sc;
      float hv = h_f[idx] + hacc[w * 256 + i];
      h_f[idx]  = hv;
      h_bf[idx] = f32_to_bf16_bits(hv);
    }
    __syncthreads();

    // readout for t+1: one 16x16x256 WMMA tile on wave 0 (N padded 10->16)
    if (w == 0) {
      v8f c = wmma_tile_pipe(robf, 0, h_bf, lane);
      if (lm < OUTD) {
        float bias = rob[lm];
#pragma unroll
        for (int r = 0; r < 8; r++) {
          int m = lh * 8 + r;
          out[((size_t)(bbase + m) * (TT + 1) + (t + 1)) * OUTD + lm] = c[r] + bias;
        }
      }
    }
  }
}

extern "C" void kernel_launch(void* const* d_in, const int* in_sizes, int n_in,
                              void* d_out, int out_size, void* d_ws, size_t ws_size,
                              hipStream_t stream) {
  const float* x0  = (const float*)d_in[0];
  const float* ls  = (const float*)d_in[1];
  const float* iW1 = (const float*)d_in[2];
  const float* ib1 = (const float*)d_in[3];
  const float* iW2 = (const float*)d_in[4];
  const float* ib2 = (const float*)d_in[5];
  const float* iW3 = (const float*)d_in[6];
  const float* ib3 = (const float*)d_in[7];
  const float* vW1 = (const float*)d_in[8];
  const float* vb1 = (const float*)d_in[9];
  const float* vW2 = (const float*)d_in[10];
  const float* vb2 = (const float*)d_in[11];
  const float* vW3 = (const float*)d_in[12];
  const float* vb3 = (const float*)d_in[13];
  const float* roW = (const float*)d_in[14];
  const float* rob = (const float*)d_in[15];
  float* out = (float*)d_out;

  // bf16 weight staging in workspace (~5 MB): W1 | W2 | W3 | ro(16x256, padded)
  unsigned short* W1bf = (unsigned short*)d_ws;
  unsigned short* W2bf = W1bf + 256 * 256;
  unsigned short* W3bf = W2bf + 256 * 256;
  unsigned short* robf = W3bf + 9216 * 256;

  cvt_bf16_kernel<<<(256 * 256 + 255) / 256, 256, 0, stream>>>(vW1, W1bf, 256 * 256);
  cvt_bf16_kernel<<<(256 * 256 + 255) / 256, 256, 0, stream>>>(vW2, W2bf, 256 * 256);
  cvt_bf16_kernel<<<(9216 * 256 + 255) / 256, 256, 0, stream>>>(vW3, W3bf, 9216 * 256);
  pad_ro_kernel<<<(16 * 256 + 255) / 256, 256, 0, stream>>>(roW, robf);

  rde_kernel<<<16, 512, 0, stream>>>(x0, ls, iW1, ib1, iW2, ib2, iW3, ib3,
                                     vb1, vb2, vb3, roW, rob,
                                     W1bf, W2bf, W3bf, robf, out);
}
